// EtcModel_77206332113557
// MI455X (gfx1250) — matrix operand order repaired and checked by reading, not compiled, not run on hardware
//
#include <hip/hip_runtime.h>
#include <hip/hip_bf16.h>
#include <math.h>

#define HIDN  768
#define HEADS 12
#define DH    64
#define BQ    2
#define LQ    4096
#define GQ    128
#define RQ    128
#define NBQ   32            // LQ / RQ
#define INTERN 3072
#define NLOC  (BQ*LQ)       // 8192
#define NGLO  (BQ*GQ)       // 256
#define GKEYS (GQ + LQ)     // 4224
#define SCALEQ 0.125f
#define NEGBIG (-1.0e9f)

typedef __attribute__((ext_vector_type(16))) __bf16 v16bf;
typedef __attribute__((ext_vector_type(8)))  float  v8f;

union Frag { v16bf v; unsigned int u[8]; };

__device__ __forceinline__ unsigned int f2bf_u(float f) {
  unsigned int u = __float_as_uint(f);
  return (u + 0x7FFFu + ((u >> 16) & 1u)) >> 16;
}
__device__ __forceinline__ unsigned short f2bf(float f) { return (unsigned short)f2bf_u(f); }
__device__ __forceinline__ unsigned int packbf(float lo, float hi) {
  return f2bf_u(lo) | (f2bf_u(hi) << 16);
}
__device__ __forceinline__ float bf2f(unsigned short h) {
  return __uint_as_float(((unsigned int)h) << 16);
}
__device__ __forceinline__ v8f wmma_bf16(const Frag& a, const Frag& b, v8f c) {
  return __builtin_amdgcn_wmma_f32_16x16x32_bf16(false, a.v, false, b.v, (short)0, c, false, false);
}
__device__ __forceinline__ float gelu_tanh(float x) {
  const float c0 = 0.7978845608028654f;
  float t = tanhf(c0 * (x + 0.044715f * x * x * x));
  return 0.5f * x * (1.0f + t);
}
__device__ __forceinline__ int rel_id_from_off(int off) {
  int oc = off < -12 ? -12 : (off > 12 ? 12 : off);
  return oc >= 0 ? oc : 12 - oc;
}
__device__ __forceinline__ float wave_sum32(float v) {
  for (int m = 16; m >= 1; m >>= 1) v += __shfl_xor(v, m, 32);
  return v;
}

// ---- CDNA5 async memory->LDS copy (ASYNCcnt path) ----
__device__ __forceinline__ void async_load_b128(unsigned lds_off, const void* gptr) {
  asm volatile("global_load_async_to_lds_b128 %0, %1, off"
               :: "v"(lds_off), "v"((unsigned long long)gptr) : "memory");
}
__device__ __forceinline__ void wait_async0() {
  asm volatile("s_wait_asynccnt 0x0" ::: "memory");
}
__device__ __forceinline__ unsigned lds_off_of(const void* p) {
  return (unsigned)(size_t)p;   // LDS aperture: low 32 bits = LDS offset
}

// -------------------- utility kernels --------------------

__global__ __launch_bounds__(256)
void convert_f32_to_bf16_k(const float* __restrict__ src, unsigned short* __restrict__ dst, int n) {
  int i = blockIdx.x * 1024 + threadIdx.x;
#pragma unroll
  for (int j = 0; j < 4; ++j) {
    int k = i + j * 256;
    if (k < n) dst[k] = f2bf(src[k]);
  }
}

__global__ __launch_bounds__(256)
void copy_out_k(const float* __restrict__ xl, const float* __restrict__ xg,
                float* __restrict__ out, size_t nl, size_t ng) {
  size_t i = (size_t)blockIdx.x * 256 + threadIdx.x;
  if (i < nl)            out[i] = xl[i];
  else if (i < nl + ng)  out[i] = xg[i - nl];
}

// -------------------- embedding + layernorm --------------------

__global__ __launch_bounds__(256)
void embed_ln_k(const float* __restrict__ tok_emb, const float* __restrict__ seg_emb,
                const float* __restrict__ g, const float* __restrict__ be,
                const int* __restrict__ tok_l, const int* __restrict__ seg_l,
                const int* __restrict__ tok_g, const int* __restrict__ seg_g,
                float* __restrict__ xl, float* __restrict__ xg,
                unsigned short* __restrict__ xlb, unsigned short* __restrict__ xgb,
                int nLocal) {
  __shared__ float buf[HIDN];
  __shared__ float rs1[8], rs2[8], mv[2];
  int row = blockIdx.x, tid = threadIdx.x;
  const int* tids; const int* sids; float* xo; unsigned short* xob; int r;
  if (row < nLocal) { tids = tok_l; sids = seg_l; xo = xl; xob = xlb; r = row; }
  else              { tids = tok_g; sids = seg_g; xo = xg; xob = xgb; r = row - nLocal; }
  int t = tids[r], sgi = sids[r];
  float s1 = 0.f, s2 = 0.f;
#pragma unroll
  for (int j = 0; j < 3; ++j) {
    int c = tid + j * 256;
    float v = tok_emb[(size_t)t * HIDN + c] + seg_emb[(size_t)sgi * HIDN + c];
    buf[c] = v; s1 += v; s2 += v * v;
  }
  s1 = wave_sum32(s1); s2 = wave_sum32(s2);
  if ((tid & 31) == 0) { rs1[tid >> 5] = s1; rs2[tid >> 5] = s2; }
  __syncthreads();
  if (tid == 0) {
    float a = 0.f, bb = 0.f;
    for (int i = 0; i < 8; ++i) { a += rs1[i]; bb += rs2[i]; }
    float mu = a / (float)HIDN;
    float var = bb / (float)HIDN - mu * mu;
    mv[0] = mu; mv[1] = rsqrtf(var + 1e-12f);
  }
  __syncthreads();
  float mu = mv[0], rstd = mv[1];
#pragma unroll
  for (int j = 0; j < 3; ++j) {
    int c = tid + j * 256;
    float v = (buf[c] - mu) * rstd * g[c] + be[c];
    xo[(size_t)r * HIDN + c] = v;
    xob[(size_t)r * HIDN + c] = f2bf(v);
  }
}

// x = LN(x + t); writes f32 in place + bf16 copy
__global__ __launch_bounds__(256)
void residual_ln_k(float* __restrict__ x, const float* __restrict__ t,
                   const float* __restrict__ g, const float* __restrict__ be,
                   unsigned short* __restrict__ xb) {
  __shared__ float buf[HIDN];
  __shared__ float rs1[8], rs2[8], mv[2];
  int row = blockIdx.x, tid = threadIdx.x;
  float s1 = 0.f, s2 = 0.f;
#pragma unroll
  for (int j = 0; j < 3; ++j) {
    int c = tid + j * 256;
    float v = x[(size_t)row * HIDN + c] + t[(size_t)row * HIDN + c];
    buf[c] = v; s1 += v; s2 += v * v;
  }
  s1 = wave_sum32(s1); s2 = wave_sum32(s2);
  if ((tid & 31) == 0) { rs1[tid >> 5] = s1; rs2[tid >> 5] = s2; }
  __syncthreads();
  if (tid == 0) {
    float a = 0.f, bb = 0.f;
    for (int i = 0; i < 8; ++i) { a += rs1[i]; bb += rs2[i]; }
    float mu = a / (float)HIDN;
    float var = bb / (float)HIDN - mu * mu;
    mv[0] = mu; mv[1] = rsqrtf(var + 1e-12f);
  }
  __syncthreads();
  float mu = mv[0], rstd = mv[1];
#pragma unroll
  for (int j = 0; j < 3; ++j) {
    int c = tid + j * 256;
    float v = (buf[c] - mu) * rstd * g[c] + be[c];
    x[(size_t)row * HIDN + c] = v;
    xb[(size_t)row * HIDN + c] = f2bf(v);
  }
}

// -------------------- generic bf16 WMMA GEMM (async double-buffered) ---------
// C[M,N] = A[M,K] * B[K,N] + bias ; A,B bf16 row-major.
// EPI: 0 = bias + f32 store, 1 = bias + bf16 store, 2 = bias + gelu + bf16 store.
// grid = (N/64, M/128), block = 256 (8 waves; 4x2 grid of 32x32 wave tiles).
// A/B tiles are streamed memory->LDS with GLOBAL_LOAD_ASYNC_TO_LDS_B128 and
// double-buffered: issue tile k+1, run WMMAs on tile k, s_wait_asynccnt 0.
template <int EPI>
__global__ __launch_bounds__(256)
void gemm_bf16_k(const unsigned short* __restrict__ A,
                 const unsigned short* __restrict__ Bw,
                 const float* __restrict__ bias,
                 float* __restrict__ outF,
                 unsigned short* __restrict__ outBF,
                 int M, int N, int K) {
  __shared__ unsigned int   sA[2][128][16];   // 2 x (128 x 32 bf16) tiles
  __shared__ unsigned short sB[2][32][64];    // 2 x (32 x 64 bf16) tiles
  const int tid  = threadIdx.x;
  const int lane = tid & 31;
  const int wave = tid >> 5;
  const int wm   = wave & 3;
  const int wn   = wave >> 2;
  const int m0   = blockIdx.y * 128;
  const int n0   = blockIdx.x * 64;

  v8f zero8 = {0.f,0.f,0.f,0.f,0.f,0.f,0.f,0.f};
  v8f acc[2][2];
  acc[0][0] = zero8; acc[0][1] = zero8; acc[1][0] = zero8; acc[1][1] = zero8;

  const int arow  = lane & 15;
  const int pbase = (lane < 16) ? 0 : 8;
  const int kb    = (lane < 16) ? 0 : 16;

  auto issue_tile = [&](int bufi, int k0) {
    // A tile: 128 rows x 32 bf16 = 8KB = 512 x 16B chunks, 2 per thread
#pragma unroll
    for (int i = 0; i < 2; ++i) {
      int ch = i * 256 + tid;
      int r = ch >> 2, part = ch & 3;
      async_load_b128(lds_off_of(&sA[bufi][r][part * 4]),
                      (const void*)(A + (size_t)(m0 + r) * K + k0 + part * 8));
    }
    // B tile: 32 rows x 64 bf16 = 4KB = 256 x 16B chunks, 1 per thread
    {
      int r = tid >> 3, part = tid & 7;
      async_load_b128(lds_off_of(&sB[bufi][r][part * 8]),
                      (const void*)(Bw + (size_t)(k0 + r) * N + n0 + part * 8));
    }
  };

  issue_tile(0, 0);
  wait_async0();
  __syncthreads();

  const int nk = K >> 5;
  for (int kt = 0; kt < nk; ++kt) {
    const int buf = kt & 1;
    if (kt + 1 < nk) issue_tile(buf ^ 1, (kt + 1) * 32);

    Frag af[2], bfr[2];
#pragma unroll
    for (int i = 0; i < 2; ++i) {
      int m = wm * 32 + i * 16 + arow;
#pragma unroll
      for (int j = 0; j < 4; ++j) {
        af[i].u[j]     = sA[buf][m][pbase + j];
        af[i].u[4 + j] = sA[buf][m][pbase + 4 + j];
      }
    }
#pragma unroll
    for (int j2 = 0; j2 < 2; ++j2) {
      int n = wn * 32 + j2 * 16 + (lane & 15);
#pragma unroll
      for (int j = 0; j < 8; ++j) {
        unsigned int lo = sB[buf][kb + 2 * j][n];
        unsigned int hi = sB[buf][kb + 2 * j + 1][n];
        bfr[j2].u[j] = lo | (hi << 16);
      }
    }
#pragma unroll
    for (int i = 0; i < 2; ++i)
#pragma unroll
      for (int j = 0; j < 2; ++j)
        acc[i][j] = wmma_bf16(af[i], bfr[j], acc[i][j]);

    wait_async0();
    __syncthreads();
  }

  const int crow = (lane < 16) ? 0 : 8;
#pragma unroll
  for (int i = 0; i < 2; ++i)
#pragma unroll
    for (int j = 0; j < 2; ++j) {
      int nn = n0 + wn * 32 + j * 16 + (lane & 15);
      float bv = bias ? bias[nn] : 0.f;
#pragma unroll
      for (int r = 0; r < 8; ++r) {
        int m = m0 + wm * 32 + i * 16 + crow + r;
        float val = acc[i][j][r] + bv;
        size_t off = (size_t)m * N + nn;
        if (EPI == 0)      outF[off]  = val;
        else if (EPI == 1) outBF[off] = f2bf(val);
        else               outBF[off] = f2bf(gelu_tanh(val));
      }
    }
}

// -------------------- local (sliding-block) attention --------------------
// grid = B * NB * 2 * HEADS, block = 256. Each WG: 64 query rows, 512 keys
// (384 local window + 128 global), rel-position terms, softmax, context.
__global__ __launch_bounds__(256)
void local_attn_k(const unsigned short* __restrict__ ql,
                  const unsigned short* __restrict__ kl,
                  const unsigned short* __restrict__ vl,
                  const unsigned short* __restrict__ kg,
                  const unsigned short* __restrict__ vg,
                  const float* __restrict__ rel,     // [32][HEADS][DH] layer slice
                  unsigned short* __restrict__ ctx)  // [B*L][HIDN]
{
  int idx = blockIdx.x;
  const int h = idx % HEADS; idx /= HEADS;
  const int s = idx & 1;     idx >>= 1;
  const int n = idx % NBQ;
  const int b = idx / NBQ;

  __shared__ unsigned int sQ[64][32];   // 64 rows x 64 bf16
  __shared__ float sS[64][512];         // scores / probs
  __shared__ float sQrel[64][26];       // q . rel[rid], rid 0..24 + l2g(25)

  const int tid  = threadIdx.x;
  const int lane = tid & 31;
  const int wave = tid >> 5;
  const int rowbase = n * RQ + s * 64;

  const unsigned int* qlu = (const unsigned int*)ql;
#pragma unroll
  for (int i = 0; i < 8; ++i) {
    int p = i * 256 + tid;               // 0..2047
    int r = p >> 5, c = p & 31;
    size_t gr = (size_t)(b * LQ + rowbase + r);
    sQ[r][c] = qlu[gr * (HIDN / 2) + h * (DH / 2) + c];
  }
  __syncthreads();

  for (int p = tid; p < 64 * 26; p += 256) {
    int r = p / 26, rid = p % 26;
    int relIdx = (rid < 25) ? rid : 25;  // L2G_REL_ID = 25
    const float* rp = rel + ((size_t)relIdx * HEADS + h) * DH;
    float acc = 0.f;
#pragma unroll
    for (int d = 0; d < DH; d += 2) {
      unsigned int pr = sQ[r][d >> 1];
      acc += bf2f((unsigned short)(pr & 0xFFFF)) * rp[d];
      acc += bf2f((unsigned short)(pr >> 16))    * rp[d + 1];
    }
    sQrel[r][rid] = acc;
  }
  __syncthreads();

  // ---- scores ----
  {
    const int rg = wave & 3;             // rows rg*16..+15
    const int kh = wave >> 2;            // keys kh*256..+255
    const int arow  = lane & 15;
    const int pbase = (lane < 16) ? 0 : 8;
    const int bkrow = lane & 15;         // key within tile
    const int bd    = (lane < 16) ? 0 : 16;
    for (int kt = 0; kt < 16; ++kt) {
      int kbidx = kh * 256 + kt * 16;
      v8f acc = {0.f,0.f,0.f,0.f,0.f,0.f,0.f,0.f};
#pragma unroll
      for (int ds = 0; ds < DH; ds += 32) {
        Frag af, bfr;
        int m = rg * 16 + arow;
        int pc = ds >> 1;
#pragma unroll
        for (int j = 0; j < 4; ++j) {
          af.u[j]     = sQ[m][pc + pbase + j];
          af.u[4 + j] = sQ[m][pc + pbase + 4 + j];
        }
        int key = kbidx + bkrow;
        const unsigned int* kptr = nullptr;
        size_t base = 0; bool ok = false;
        if (key < 3 * RQ) {
          int pp = n * RQ - RQ + key;
          if (pp >= 0 && pp < LQ) {
            kptr = (const unsigned int*)kl;
            base = ((size_t)(b * LQ + pp) * HIDN + h * DH + ds + bd) >> 1; ok = true;
          }
        } else {
          kptr = (const unsigned int*)kg;
          base = ((size_t)(b * GQ + key - 3 * RQ) * HIDN + h * DH + ds + bd) >> 1; ok = true;
        }
        if (ok) {
#pragma unroll
          for (int j = 0; j < 8; ++j) bfr.u[j] = kptr[base + j];
        } else {
#pragma unroll
          for (int j = 0; j < 8; ++j) bfr.u[j] = 0u;
        }
        acc = wmma_bf16(af, bfr, acc);
      }
      const int crow = (lane < 16) ? 0 : 8;
      const int k = kbidx + (lane & 15);
#pragma unroll
      for (int r = 0; r < 8; ++r) {
        int qloc = rg * 16 + crow + r;
        float sc;
        if (k < 3 * RQ) {
          int qib = s * 64 + qloc;
          int off = k - RQ - qib;
          int pp  = n * RQ - RQ + k;
          bool valid = (off >= -RQ && off <= RQ) && (pp >= 0 && pp < LQ);
          sc = valid ? (acc[r] + sQrel[qloc][rel_id_from_off(off)]) * SCALEQ
                     : NEGBIG * SCALEQ;
        } else {
          sc = (acc[r] + sQrel[qloc][25]) * SCALEQ;
        }
        sS[qloc][k] = sc;
      }
    }
  }
  __syncthreads();

  // ---- softmax (each wave: 8 rows, lanes strided over 512) ----
  for (int rr = 0; rr < 8; ++rr) {
    int q = wave * 8 + rr;
    float mx = -3.4e38f;
#pragma unroll
    for (int i = 0; i < 16; ++i) mx = fmaxf(mx, sS[q][lane + i * 32]);
    for (int m2 = 16; m2 >= 1; m2 >>= 1) mx = fmaxf(mx, __shfl_xor(mx, m2, 32));
    float sum = 0.f;
#pragma unroll
    for (int i = 0; i < 16; ++i) {
      float e = __expf(sS[q][lane + i * 32] - mx);
      sS[q][lane + i * 32] = e; sum += e;
    }
    sum = wave_sum32(sum);
    float inv = 1.0f / sum;
#pragma unroll
    for (int i = 0; i < 16; ++i) sS[q][lane + i * 32] *= inv;
  }
  __syncthreads();

  // ---- context: P(64x512) @ Vcat(512x64) ----
  {
    const int rg = wave & 3;            // rows rg*16
    const int ch = wave >> 2;           // cols ch*32 (2 tiles)
    const int arow = lane & 15;
    const int koff = (lane < 16) ? 0 : 16;
    const int kb2  = (lane < 16) ? 0 : 16;
    v8f acc0 = {0.f,0.f,0.f,0.f,0.f,0.f,0.f,0.f};
    v8f acc1 = acc0;
    for (int ks = 0; ks < 16; ++ks) {
      int kbase = ks * 32;
      Frag af;
      int q = rg * 16 + arow;
#pragma unroll
      for (int j = 0; j < 4; ++j) {
        af.u[j]     = packbf(sS[q][kbase + koff + 2 * j],      sS[q][kbase + koff + 2 * j + 1]);
        af.u[4 + j] = packbf(sS[q][kbase + koff + 8 + 2 * j], sS[q][kbase + koff + 8 + 2 * j + 1]);
      }
#pragma unroll
      for (int jt = 0; jt < 2; ++jt) {
        Frag bfr;
        int dcol = h * DH + ch * 32 + jt * 16 + (lane & 15);
#pragma unroll
        for (int j = 0; j < 8; ++j) {
          int k0i = kbase + kb2 + 2 * j;
          unsigned short lo, hi;
          if (k0i < 3 * RQ) {
            int p0 = n * RQ - RQ + k0i;
            lo = (p0 >= 0 && p0 < LQ)     ? vl[(size_t)(b * LQ + p0)     * HIDN + dcol] : (unsigned short)0;
            hi = (p0 + 1 >= 0 && p0 + 1 < LQ) ? vl[(size_t)(b * LQ + p0 + 1) * HIDN + dcol] : (unsigned short)0;
          } else {
            int j0 = k0i - 3 * RQ;
            lo = vg[(size_t)(b * GQ + j0)     * HIDN + dcol];
            hi = vg[(size_t)(b * GQ + j0 + 1) * HIDN + dcol];
          }
          bfr.u[j] = (unsigned int)lo | ((unsigned int)hi << 16);
        }
        if (jt == 0) acc0 = wmma_bf16(af, bfr, acc0);
        else         acc1 = wmma_bf16(af, bfr, acc1);
      }
    }
    const int crow = (lane < 16) ? 0 : 8;
#pragma unroll
    for (int r = 0; r < 8; ++r) {
      int qloc = rg * 16 + crow + r;
      size_t gr = (size_t)(b * LQ + rowbase + qloc);
      int c0 = h * DH + ch * 32 + (lane & 15);
      ctx[gr * HIDN + c0]      = f2bf(acc0[r]);
      ctx[gr * HIDN + c0 + 16] = f2bf(acc1[r]);
    }
  }
}

// -------------------- global attention: scores --------------------
// grid = B*HEADS, block = 256. scores[b][h][128][GKEYS] into ws.
__global__ __launch_bounds__(256)
void global_score_k(const unsigned short* __restrict__ qg,
                    const unsigned short* __restrict__ kg,
                    const unsigned short* __restrict__ kl,
                    const float* __restrict__ rel,
                    float* __restrict__ scores) {
  const int h = blockIdx.x % HEADS;
  const int b = blockIdx.x / HEADS;
  __shared__ unsigned int sQ[128][32];
  __shared__ float sQrel[128][26];
  const int tid  = threadIdx.x;
  const int lane = tid & 31;
  const int wave = tid >> 5;

  const unsigned int* qgu = (const unsigned int*)qg;
#pragma unroll
  for (int i = 0; i < 16; ++i) {
    int p = i * 256 + tid;               // 0..4095
    int r = p >> 5, c = p & 31;
    sQ[r][c] = qgu[(size_t)(b * GQ + r) * (HIDN / 2) + h * (DH / 2) + c];
  }
  __syncthreads();
  for (int p = tid; p < 128 * 26; p += 256) {
    int r = p / 26, rid = p % 26;
    int relIdx = (rid < 25) ? rid : 26;  // G2L_REL_ID = 26
    const float* rp = rel + ((size_t)relIdx * HEADS + h) * DH;
    float acc = 0.f;
#pragma unroll
    for (int d = 0; d < DH; d += 2) {
      unsigned int pr = sQ[r][d >> 1];
      acc += bf2f((unsigned short)(pr & 0xFFFF)) * rp[d];
      acc += bf2f((unsigned short)(pr >> 16))    * rp[d + 1];
    }
    sQrel[r][rid] = acc;
  }
  __syncthreads();

  const int arow  = lane & 15;
  const int pbase = (lane < 16) ? 0 : 8;
  const int bkrow = lane & 15;
  const int bd    = (lane < 16) ? 0 : 16;
  float* out = scores + ((size_t)b * HEADS + h) * GQ * GKEYS;

  for (int kt = 0; kt < GKEYS / 16; ++kt) {   // 264
    int kbidx = kt * 16;
    v8f acc = {0.f,0.f,0.f,0.f,0.f,0.f,0.f,0.f};
#pragma unroll
    for (int ds = 0; ds < DH; ds += 32) {
      Frag af, bfr;
      int m = wave * 16 + arow;
      int pc = ds >> 1;
#pragma unroll
      for (int j = 0; j < 4; ++j) {
        af.u[j]     = sQ[m][pc + pbase + j];
        af.u[4 + j] = sQ[m][pc + pbase + 4 + j];
      }
      int key = kbidx + bkrow;
      const unsigned int* kptr;
      size_t base;
      if (key < GQ) {
        kptr = (const unsigned int*)kg;
        base = ((size_t)(b * GQ + key) * HIDN + h * DH + ds + bd) >> 1;
      } else {
        kptr = (const unsigned int*)kl;
        base = ((size_t)(b * LQ + key - GQ) * HIDN + h * DH + ds + bd) >> 1;
      }
      // stream next key tile into L2 while we do the WMMAs
      __builtin_prefetch((const void*)(kptr + base + (16 * HIDN / 2)), 0, 3);
#pragma unroll
      for (int j = 0; j < 8; ++j) bfr.u[j] = kptr[base + j];
      acc = wmma_bf16(af, bfr, acc);
    }
    const int crow = (lane < 16) ? 0 : 8;
    const int k = kbidx + (lane & 15);
#pragma unroll
    for (int r = 0; r < 8; ++r) {
      int q = wave * 16 + crow + r;
      float sc;
      if (k < GQ) sc = (acc[r] + sQrel[q][rel_id_from_off(k - q)]) * SCALEQ;
      else        sc = (acc[r] + sQrel[q][25]) * SCALEQ;
      out[(size_t)q * GKEYS + k] = sc;
    }
  }
}

// one block per row, softmax over ncols in place
__global__ __launch_bounds__(256)
void softmax_rows_k(float* __restrict__ sc, int ncols) {
  __shared__ float red[8];
  __shared__ float bc[1];
  size_t row = blockIdx.x;
  float* p = sc + row * (size_t)ncols;
  const int tid = threadIdx.x, lane = tid & 31, wave = tid >> 5;
  float mx = -3.4e38f;
  for (int c = tid; c < ncols; c += 256) mx = fmaxf(mx, p[c]);
  for (int m2 = 16; m2 >= 1; m2 >>= 1) mx = fmaxf(mx, __shfl_xor(mx, m2, 32));
  if (lane == 0) red[wave] = mx;
  __syncthreads();
  if (tid == 0) {
    float m = red[0];
    for (int i = 1; i < 8; ++i) m = fmaxf(m, red[i]);
    bc[0] = m;
  }
  __syncthreads();
  mx = bc[0];
  __syncthreads();
  float sum = 0.f;
  for (int c = tid; c < ncols; c += 256) {
    float e = __expf(p[c] - mx);
    p[c] = e; sum += e;
  }
  sum = wave_sum32(sum);
  if (lane == 0) red[wave] = sum;
  __syncthreads();
  if (tid == 0) {
    float a = 0.f;
    for (int i = 0; i < 8; ++i) a += red[i];
    bc[0] = 1.0f / a;
  }
  __syncthreads();
  float inv = bc[0];
  for (int c = tid; c < ncols; c += 256) p[c] *= inv;
}

// grid = B*HEADS : ctx_g(128x64) = P(128xGKEYS) @ Vcat(GKEYSx64)
__global__ __launch_bounds__(256)
void global_ctx_k(const float* __restrict__ scores,
                  const unsigned short* __restrict__ vg,
                  const unsigned short* __restrict__ vl,
                  unsigned short* __restrict__ ctxg) {
  const int h = blockIdx.x % HEADS;
  const int b = blockIdx.x / HEADS;
  const int tid  = threadIdx.x;
  const int lane = tid & 31;
  const int wave = tid >> 5;
  const float* P = scores + ((size_t)b * HEADS + h) * GQ * GKEYS;

  const int arow = lane & 15;
  const int koff = (lane < 16) ? 0 : 16;
  const int kb2  = (lane < 16) ? 0 : 16;
  v8f zero8 = {0.f,0.f,0.f,0.f,0.f,0.f,0.f,0.f};
  v8f acc[4];
  acc[0] = zero8; acc[1] = zero8; acc[2] = zero8; acc[3] = zero8;

  for (int ks = 0; ks < GKEYS / 32; ++ks) {   // 132
    int kbase = ks * 32;
    Frag af;
    int q = wave * 16 + arow;
    const float* pr = P + (size_t)q * GKEYS + kbase;
    __builtin_prefetch((const void*)(pr + 32), 0, 3);
#pragma unroll
    for (int j = 0; j < 4; ++j) {
      af.u[j]     = packbf(pr[koff + 2 * j],      pr[koff + 2 * j + 1]);
      af.u[4 + j] = packbf(pr[koff + 8 + 2 * j], pr[koff + 8 + 2 * j + 1]);
    }
#pragma unroll
    for (int jt = 0; jt < 4; ++jt) {
      Frag bfr;
      int dcol = h * DH + jt * 16 + (lane & 15);
#pragma unroll
      for (int j = 0; j < 8; ++j) {
        int key = kbase + kb2 + 2 * j;
        unsigned short lo, hi;
        if (key < GQ) {
          lo = vg[(size_t)(b * GQ + key)     * HIDN + dcol];
          hi = vg[(size_t)(b * GQ + key + 1) * HIDN + dcol];
        } else {
          lo = vl[(size_t)(b * LQ + key - GQ)     * HIDN + dcol];
          hi = vl[(size_t)(b * LQ + key - GQ + 1) * HIDN + dcol];
        }
        bfr.u[j] = (unsigned int)lo | ((unsigned int)hi << 16);
      }
      acc[jt] = wmma_bf16(af, bfr, acc[jt]);
    }
  }
  const int crow = (lane < 16) ? 0 : 8;
#pragma unroll
  for (int jt = 0; jt < 4; ++jt)
#pragma unroll
    for (int r = 0; r < 8; ++r) {
      int q = wave * 16 + crow + r;
      size_t gr = (size_t)(b * GQ + q);
      int c0 = h * DH + jt * 16 + (lane & 15);
      ctxg[gr * HIDN + c0] = f2bf(acc[jt][r]);
    }
}

// -------------------- host orchestration --------------------

extern "C" void kernel_launch(void* const* d_in, const int* in_sizes, int n_in,
                              void* d_out, int out_size, void* d_ws, size_t ws_size,
                              hipStream_t stream) {
  (void)in_sizes; (void)n_in; (void)out_size; (void)ws_size;
  const float* tok_emb = (const float*)d_in[0];
  const float* seg_emb = (const float*)d_in[1];
  const float* emb_ln_g = (const float*)d_in[2];
  const float* emb_ln_b = (const float*)d_in[3];
  const float* wq  = (const float*)d_in[4];
  const float* bq  = (const float*)d_in[5];
  const float* wk  = (const float*)d_in[6];
  const float* bk  = (const float*)d_in[7];
  const float* wv  = (const float*)d_in[8];
  const float* bv  = (const float*)d_in[9];
  const float* rel = (const float*)d_in[10];
  const float* wol = (const float*)d_in[11];
  const float* bol = (const float*)d_in[12];
  const float* wog = (const float*)d_in[13];
  const float* bog = (const float*)d_in[14];
  const float* ln1lg = (const float*)d_in[15];
  const float* ln1lb = (const float*)d_in[16];
  const float* ln1gg = (const float*)d_in[17];
  const float* ln1gb = (const float*)d_in[18];
  const float* ln2lg = (const float*)d_in[19];
  const float* ln2lb = (const float*)d_in[20];
  const float* ln2gg = (const float*)d_in[21];
  const float* ln2gb = (const float*)d_in[22];
  const float* w1 = (const float*)d_in[23];
  const float* b1 = (const float*)d_in[24];
  const float* w2 = (const float*)d_in[25];
  const float* b2 = (const float*)d_in[26];
  const int* tok_l = (const int*)d_in[27];
  const int* seg_l = (const int*)d_in[28];
  const int* tok_g = (const int*)d_in[29];
  const int* seg_g = (const int*)d_in[30];

  char* wp = (char*)d_ws;
  auto alloc = [&](size_t bytes) -> void* {
    void* r = (void*)wp;
    wp += (bytes + 255) & ~(size_t)255;
    return r;
  };
  float* xl = (float*)alloc((size_t)NLOC * HIDN * 4);
  float* xg = (float*)alloc((size_t)NGLO * HIDN * 4);
  unsigned short* xlb = (unsigned short*)alloc((size_t)NLOC * HIDN * 2);
  unsigned short* xgb = (unsigned short*)alloc((size_t)NGLO * HIDN * 2);
  unsigned short* qlb = (unsigned short*)alloc((size_t)NLOC * HIDN * 2);
  unsigned short* klb = (unsigned short*)alloc((size_t)NLOC * HIDN * 2);
  unsigned short* vlb = (unsigned short*)alloc((size_t)NLOC * HIDN * 2);
  unsigned short* qgb = (unsigned short*)alloc((size_t)NGLO * HIDN * 2);
  unsigned short* kgb = (unsigned short*)alloc((size_t)NGLO * HIDN * 2);
  unsigned short* vgb = (unsigned short*)alloc((size_t)NGLO * HIDN * 2);
  unsigned short* ctxl = (unsigned short*)alloc((size_t)NLOC * HIDN * 2);
  unsigned short* ctxg = (unsigned short*)alloc((size_t)NGLO * HIDN * 2);
  float* tmpl = (float*)alloc((size_t)NLOC * HIDN * 4);
  float* tmpg = (float*)alloc((size_t)NGLO * HIDN * 4);
  // BIG region: reused as (a) global attention scores, (b) FFN hidden (bf16)
  char* big = (char*)alloc((size_t)BQ * HEADS * GQ * GKEYS * 4);   // 51,904,512 B
  float* scoresg = (float*)big;
  unsigned short* hl = (unsigned short*)big;                         // NLOC*INTERN bf16
  unsigned short* hg = (unsigned short*)(big + (size_t)NLOC * INTERN * 2);
  // per-layer bf16 weights
  unsigned short* wbq  = (unsigned short*)alloc((size_t)HIDN * HIDN * 2);
  unsigned short* wbk  = (unsigned short*)alloc((size_t)HIDN * HIDN * 2);
  unsigned short* wbv  = (unsigned short*)alloc((size_t)HIDN * HIDN * 2);
  unsigned short* wbol = (unsigned short*)alloc((size_t)HIDN * HIDN * 2);
  unsigned short* wbog = (unsigned short*)alloc((size_t)HIDN * HIDN * 2);
  unsigned short* wb1  = (unsigned short*)alloc((size_t)HIDN * INTERN * 2);
  unsigned short* wb2  = (unsigned short*)alloc((size_t)INTERN * HIDN * 2);

  auto conv = [&](const float* src, unsigned short* dst, int n) {
    convert_f32_to_bf16_k<<<dim3((n + 1023) / 1024), dim3(256), 0, stream>>>(src, dst, n);
  };
  auto gemm = [&](int epi, const unsigned short* A, const unsigned short* Bm, const float* bias,
                  float* of, unsigned short* ob, int M, int N, int K) {
    dim3 g(N / 64, M / 128), blk(256);
    if (epi == 0)      gemm_bf16_k<0><<<g, blk, 0, stream>>>(A, Bm, bias, of, ob, M, N, K);
    else if (epi == 1) gemm_bf16_k<1><<<g, blk, 0, stream>>>(A, Bm, bias, of, ob, M, N, K);
    else               gemm_bf16_k<2><<<g, blk, 0, stream>>>(A, Bm, bias, of, ob, M, N, K);
  };

  // embeddings + LN
  embed_ln_k<<<dim3(NLOC + NGLO), dim3(256), 0, stream>>>(
      tok_emb, seg_emb, emb_ln_g, emb_ln_b, tok_l, seg_l, tok_g, seg_g,
      xl, xg, xlb, xgb, NLOC);

  for (int l = 0; l < 12; ++l) {
    size_t o768 = (size_t)l * HIDN * HIDN;
    size_t ow1  = (size_t)l * HIDN * INTERN;
    const float* relL = rel + (size_t)l * 32 * HEADS * DH;
    conv(wq + o768,  wbq,  HIDN * HIDN);
    conv(wk + o768,  wbk,  HIDN * HIDN);
    conv(wv + o768,  wbv,  HIDN * HIDN);
    conv(wol + o768, wbol, HIDN * HIDN);
    conv(wog + o768, wbog, HIDN * HIDN);
    conv(w1 + ow1,   wb1,  HIDN * INTERN);
    conv(w2 + ow1,   wb2,  INTERN * HIDN);

    // QKV projections
    gemm(1, xlb, wbq, bq + (size_t)l * HIDN, nullptr, qlb, NLOC, HIDN, HIDN);
    gemm(1, xlb, wbk, bk + (size_t)l * HIDN, nullptr, klb, NLOC, HIDN, HIDN);
    gemm(1, xlb, wbv, bv + (size_t)l * HIDN, nullptr, vlb, NLOC, HIDN, HIDN);
    gemm(1, xgb, wbq, bq + (size_t)l * HIDN, nullptr, qgb, NGLO, HIDN, HIDN);
    gemm(1, xgb, wbk, bk + (size_t)l * HIDN, nullptr, kgb, NGLO, HIDN, HIDN);
    gemm(1, xgb, wbv, bv + (size_t)l * HIDN, nullptr, vgb, NGLO, HIDN, HIDN);

    // local attention
    local_attn_k<<<dim3(BQ * NBQ * 2 * HEADS), dim3(256), 0, stream>>>(
        qlb, klb, vlb, kgb, vgb, relL, ctxl);

    // global attention (scores in ws -> softmax -> context)
    global_score_k<<<dim3(BQ * HEADS), dim3(256), 0, stream>>>(qgb, kgb, klb, relL, scoresg);
    softmax_rows_k<<<dim3(BQ * HEADS * GQ), dim3(256), 0, stream>>>(scoresg, GKEYS);
    global_ctx_k<<<dim3(BQ * HEADS), dim3(256), 0, stream>>>(scoresg, vgb, vlb, ctxg);

    // output projections + residual LN
    gemm(0, ctxl, wbol, bol + (size_t)l * HIDN, tmpl, nullptr, NLOC, HIDN, HIDN);
    residual_ln_k<<<dim3(NLOC), dim3(256), 0, stream>>>(
        xl, tmpl, ln1lg + (size_t)l * HIDN, ln1lb + (size_t)l * HIDN, xlb);
    gemm(0, ctxg, wbog, bog + (size_t)l * HIDN, tmpg, nullptr, NGLO, HIDN, HIDN);
    residual_ln_k<<<dim3(NGLO), dim3(256), 0, stream>>>(
        xg, tmpg, ln1gg + (size_t)l * HIDN, ln1gb + (size_t)l * HIDN, xgb);

    // FFN local
    gemm(2, xlb, wb1, b1 + (size_t)l * INTERN, nullptr, hl, NLOC, INTERN, HIDN);
    gemm(0, hl,  wb2, b2 + (size_t)l * HIDN,   tmpl, nullptr, NLOC, HIDN, INTERN);
    residual_ln_k<<<dim3(NLOC), dim3(256), 0, stream>>>(
        xl, tmpl, ln2lg + (size_t)l * HIDN, ln2lb + (size_t)l * HIDN, xlb);
    // FFN global
    gemm(2, xgb, wb1, b1 + (size_t)l * INTERN, nullptr, hg, NGLO, INTERN, HIDN);
    gemm(0, hg,  wb2, b2 + (size_t)l * HIDN,   tmpg, nullptr, NGLO, HIDN, INTERN);
    residual_ln_k<<<dim3(NGLO), dim3(256), 0, stream>>>(
        xg, tmpg, ln2gg + (size_t)l * HIDN, ln2gb + (size_t)l * HIDN, xgb);
  }

  size_t nl = (size_t)NLOC * HIDN, ng = (size_t)NGLO * HIDN;
  copy_out_k<<<dim3((unsigned)((nl + ng + 255) / 256)), dim3(256), 0, stream>>>(
      xl, xg, (float*)d_out, nl, ng);
}
